// LlamaAttention_75703093559479
// MI455X (gfx1250) — matrix operand order repaired and checked by reading, not compiled, not run on hardware
//
#include <hip/hip_runtime.h>
#include <hip/hip_bf16.h>

// Problem constants (from reference)
#define B_  2
#define S_  2048
#define H_  2048
#define NH_ 16
#define HD_ 128
#define M_  (B_*S_)   // 4096 rows total

typedef __attribute__((ext_vector_type(16))) __bf16 v16bf;
typedef __attribute__((ext_vector_type(8)))  float  v8f;

#define WMMA_BF16(a, b, c) \
  __builtin_amdgcn_wmma_f32_16x16x32_bf16(false, (a), false, (b), (short)0, (c), false, false)

// ---- Fragment loaders (CDNA5 wave32 WMMA layouts, CDNA5 ISA §7.12.2) ----
// A 16x32 (16-bit): lanes 0-15 row M=lane, comps0-7=K kb..kb+7, comps8-15=K kb+16..kb+23
// where kb = (lane>>4)*8.
static __device__ __forceinline__ v16bf load_a_f32(const float* base, int ld, int k0) {
  const int lane = threadIdx.x & 31;
  const float* p = base + (size_t)(lane & 15) * ld + k0 + ((lane >> 4) << 3);
  v16bf a;
#pragma unroll
  for (int i = 0; i < 8; ++i) a[i]     = (__bf16)p[i];
#pragma unroll
  for (int i = 0; i < 8; ++i) a[i + 8] = (__bf16)p[16 + i];
  return a;
}
static __device__ __forceinline__ v16bf load_a_bf16(const __bf16* base, int ld, int k0) {
  const int lane = threadIdx.x & 31;
  const __bf16* p = base + (size_t)(lane & 15) * ld + k0 + ((lane >> 4) << 3);
  v16bf a;
#pragma unroll
  for (int i = 0; i < 8; ++i) a[i]     = p[i];
#pragma unroll
  for (int i = 0; i < 8; ++i) a[i + 8] = p[16 + i];
  return a;
}
// B 32x16 (16-bit): lane column N=lane&15, comps0-7=K kb..kb+7, comps8-15=K kb+16..kb+23, kb=(lane>>4)*8.
// Source is W^T access: column n over k is row n of row-major W -> contiguous loads.
static __device__ __forceinline__ v16bf load_bT_f32(const float* base, int ld, int k0) {
  const int lane = threadIdx.x & 31;
  const float* p = base + (size_t)(lane & 15) * ld + k0 + ((lane >> 4) << 3);
  v16bf b;
#pragma unroll
  for (int i = 0; i < 8; ++i) b[i] = (__bf16)p[i];
#pragma unroll
  for (int i = 0; i < 8; ++i) b[i + 8] = (__bf16)p[16 + i];
  return b;
}
static __device__ __forceinline__ v16bf load_bT_bf16(const __bf16* base, int ld, int k0) {
  const int lane = threadIdx.x & 31;
  const __bf16* p = base + (size_t)(lane & 15) * ld + k0 + ((lane >> 4) << 3);
  v16bf b;
#pragma unroll
  for (int i = 0; i < 8; ++i) b[i] = p[i];
#pragma unroll
  for (int i = 0; i < 8; ++i) b[i + 8] = p[16 + i];
  return b;
}

// ================= Kernel 1: fused QKV projection + RoPE =================
// Grid (M/64, NH), block 256 (8 waves). Wave w: rows m0+(w>>2)*32, head column
// pair (d, d+64) with d in [ (w&3)*16, +16 ) so RoPE halves stay in-wave.
__global__ __launch_bounds__(256)
void qkv_rope_kernel(const float* __restrict__ hs,
                     const float* __restrict__ wq,
                     const float* __restrict__ wk,
                     const float* __restrict__ wv,
                     __bf16* __restrict__ qo,   // [B,NH,S,HD]
                     __bf16* __restrict__ ko,   // [B,NH,S,HD]
                     __bf16* __restrict__ vo) { // [B,NH,HD,S] (transposed)
  const int w    = threadIdx.x >> 5;
  const int lane = threadIdx.x & 31;
  const int h    = blockIdx.y;
  const int m0   = blockIdx.x * 64 + (w >> 2) * 32;
  const int ti   = w & 3;
  const int n1   = h * HD_ + ti * 16;   // columns d   .. d+15   (d < 64)
  const int n2   = n1 + 64;             // columns d+64.. d+79

  v8f cq[2][2] = {}, ck[2][2] = {}, cv[2][2] = {};

  for (int k0 = 0; k0 < H_; k0 += 32) {
    v16bf a0 = load_a_f32(hs + (size_t)m0        * H_, H_, k0);
    v16bf a1 = load_a_f32(hs + (size_t)(m0 + 16) * H_, H_, k0);
    {
      v16bf b0 = load_bT_f32(wq + (size_t)n1 * H_, H_, k0);
      v16bf b1 = load_bT_f32(wq + (size_t)n2 * H_, H_, k0);
      cq[0][0] = WMMA_BF16(a0, b0, cq[0][0]);
      cq[0][1] = WMMA_BF16(a0, b1, cq[0][1]);
      cq[1][0] = WMMA_BF16(a1, b0, cq[1][0]);
      cq[1][1] = WMMA_BF16(a1, b1, cq[1][1]);
    }
    {
      v16bf b0 = load_bT_f32(wk + (size_t)n1 * H_, H_, k0);
      v16bf b1 = load_bT_f32(wk + (size_t)n2 * H_, H_, k0);
      ck[0][0] = WMMA_BF16(a0, b0, ck[0][0]);
      ck[0][1] = WMMA_BF16(a0, b1, ck[0][1]);
      ck[1][0] = WMMA_BF16(a1, b0, ck[1][0]);
      ck[1][1] = WMMA_BF16(a1, b1, ck[1][1]);
    }
    {
      v16bf b0 = load_bT_f32(wv + (size_t)n1 * H_, H_, k0);
      v16bf b1 = load_bT_f32(wv + (size_t)n2 * H_, H_, k0);
      cv[0][0] = WMMA_BF16(a0, b0, cv[0][0]);
      cv[0][1] = WMMA_BF16(a0, b1, cv[0][1]);
      cv[1][0] = WMMA_BF16(a1, b0, cv[1][0]);
      cv[1][1] = WMMA_BF16(a1, b1, cv[1][1]);
    }
  }

  // RoPE + store. inv_freq depends only on lane's column d.
  const int   col = lane & 15;
  const int   d   = ti * 16 + col;                      // 0..63
  const float inv_freq = exp2f(-(float)d * (1.0f / 64.0f) * 13.287712379549449f); // log2(10000)

#pragma unroll
  for (int mi = 0; mi < 2; ++mi) {
#pragma unroll
    for (int r = 0; r < 8; ++r) {
      const int m = m0 + mi * 16 + ((lane >> 4) << 3) + r;
      const int b = m >> 11;            // m / S_
      const int s = m & (S_ - 1);
      const float ang = (float)s * inv_freq;
      const float cs = cosf(ang), sn = sinf(ang);
      const float q1 = cq[mi][0][r], q2 = cq[mi][1][r];
      const float k1 = ck[mi][0][r], k2 = ck[mi][1][r];
      const size_t base  = ((size_t)(b * NH_ + h) * S_ + s) * HD_;
      qo[base + d]      = (__bf16)(q1 * cs - q2 * sn);
      qo[base + d + 64] = (__bf16)(q2 * cs + q1 * sn);
      ko[base + d]      = (__bf16)(k1 * cs - k2 * sn);
      ko[base + d + 64] = (__bf16)(k2 * cs + k1 * sn);
      const size_t vbase = (size_t)(b * NH_ + h) * HD_ * S_;
      vo[vbase + (size_t)d        * S_ + s] = (__bf16)cv[mi][0][r];
      vo[vbase + (size_t)(d + 64) * S_ + s] = (__bf16)cv[mi][1][r];
    }
  }
}

// ================= Kernel 2: flash attention =================
// Grid (S/128, NH, B), block 256. Wave w owns 16 q-rows, streams keys in
// steps of 32 with online softmax. P goes through per-wave LDS (C->A layout).
__global__ __launch_bounds__(256)
void flash_attn_kernel(const __bf16* __restrict__ q,
                       const __bf16* __restrict__ k,
                       const __bf16* __restrict__ vT,
                       const int*    __restrict__ amask,
                       __bf16* __restrict__ out) { // [B,S,H]
  __shared__ __bf16 lds_p[8][16][32];   // 8 KB, per-wave private tiles
  const int w    = threadIdx.x >> 5;
  const int lane = threadIdx.x & 31;
  const int b    = blockIdx.z, h = blockIdx.y;
  const int q0   = blockIdx.x * 128 + w * 16;

  const __bf16* qb = q  + (size_t)(b * NH_ + h) * S_  * HD_;
  const __bf16* kb = k  + (size_t)(b * NH_ + h) * S_  * HD_;
  const __bf16* vb = vT + (size_t)(b * NH_ + h) * HD_ * S_;
  const int*    mk = amask + b * S_;

  // Resident Q fragments: 16 rows x 128 K, 4 A-frags.
  v16bf aq[4];
#pragma unroll
  for (int kk = 0; kk < 4; ++kk) aq[kk] = load_a_bf16(qb + (size_t)q0 * HD_, HD_, kk * 32);

  v8f o[8] = {};                 // 16 rows x 128 d (8 d-tiles)
  float mrow[8], lrow[8];
#pragma unroll
  for (int r = 0; r < 8; ++r) { mrow[r] = -1e30f; lrow[r] = 0.0f; }

  const float scale = 0.08838834764831845f;   // 1/sqrt(HD)

  for (int j0 = 0; j0 < S_; j0 += 32) {
    if (j0 + 32 < S_) {   // prefetch next K/V tiles (global_prefetch_b8)
      __builtin_prefetch(kb + (size_t)(j0 + 32 + (lane & 15)) * HD_, 0, 1);
      __builtin_prefetch(vb + (size_t)(lane & 15) * S_ + j0 + 32, 0, 1);
    }
    // ---- scores: two 16x16 tiles over key columns j0..j0+31 ----
    v8f st[2];
#pragma unroll
    for (int t = 0; t < 2; ++t) {
      v8f sc = {};
#pragma unroll
      for (int kk = 0; kk < 4; ++kk) {
        v16bf bk = load_bT_bf16(kb + (size_t)(j0 + t * 16) * HD_, HD_, kk * 32);
        sc = WMMA_BF16(aq[kk], bk, sc);
      }
      const int jc = j0 + t * 16 + (lane & 15);
      const float bias = (mk[jc] == 0) ? -1e30f : 0.0f;
#pragma unroll
      for (int r = 0; r < 8; ++r) sc[r] = sc[r] * scale + bias;
      st[t] = sc;
    }
    // ---- online softmax: row reductions across 16-lane column groups ----
    float al[8];
#pragma unroll
    for (int r = 0; r < 8; ++r) {
      float v = fmaxf(st[0][r], st[1][r]);
      v = fmaxf(v, __shfl_xor(v, 1, 32));
      v = fmaxf(v, __shfl_xor(v, 2, 32));
      v = fmaxf(v, __shfl_xor(v, 4, 32));
      v = fmaxf(v, __shfl_xor(v, 8, 32));
      const float mnew = fmaxf(mrow[r], v);
      al[r] = __expf(mrow[r] - mnew);
      mrow[r] = mnew;
    }
#pragma unroll
    for (int r = 0; r < 8; ++r) {
      const float p0 = __expf(st[0][r] - mrow[r]);
      const float p1 = __expf(st[1][r] - mrow[r]);
      st[0][r] = p0; st[1][r] = p1;
      float ps = p0 + p1;
      ps += __shfl_xor(ps, 1, 32);
      ps += __shfl_xor(ps, 2, 32);
      ps += __shfl_xor(ps, 4, 32);
      ps += __shfl_xor(ps, 8, 32);
      lrow[r] = lrow[r] * al[r] + ps;
    }
#pragma unroll
    for (int dt = 0; dt < 8; ++dt)
#pragma unroll
      for (int r = 0; r < 8; ++r) o[dt][r] *= al[r];

    // ---- P (C-layout) -> LDS -> A-layout bf16 fragment ----
#pragma unroll
    for (int r = 0; r < 8; ++r) {
      const int row = ((lane >> 4) << 3) + r;
      lds_p[w][row][lane & 15]        = (__bf16)st[0][r];
      lds_p[w][row][16 + (lane & 15)] = (__bf16)st[1][r];
    }
    __syncthreads();
    v16bf ap;
    {
      const int row = lane & 15;
      const int kb2 = (lane >> 4) << 3;
#pragma unroll
      for (int i = 0; i < 8; ++i) ap[i]     = lds_p[w][row][kb2 + i];
#pragma unroll
      for (int i = 0; i < 8; ++i) ap[i + 8] = lds_p[w][row][kb2 + 16 + i];
    }
    // ---- O += P @ V  (V^T gives contiguous B columns) ----
#pragma unroll
    for (int dt = 0; dt < 8; ++dt) {
      v16bf bv = load_bT_bf16(vb + (size_t)(dt * 16) * S_, S_, j0);
      o[dt] = WMMA_BF16(ap, bv, o[dt]);
    }
    __syncthreads();
  }

  // ---- normalize and store bf16 to [B,S,H] ----
  float inv_l[8];
#pragma unroll
  for (int r = 0; r < 8; ++r) inv_l[r] = 1.0f / lrow[r];
#pragma unroll
  for (int dt = 0; dt < 8; ++dt) {
#pragma unroll
    for (int r = 0; r < 8; ++r) {
      const int s = q0 + ((lane >> 4) << 3) + r;
      const size_t idx = ((size_t)(b * S_ + s)) * H_ + h * HD_ + dt * 16 + (lane & 15);
      out[idx] = (__bf16)(o[dt][r] * inv_l[r]);
    }
  }
}

// ================= Kernel 3: output projection =================
// Grid (M/64, H/128), block 256. Wave = 32x32 output tile (2x2 wmma).
__global__ __launch_bounds__(256)
void out_proj_kernel(const __bf16* __restrict__ a,   // [M, H] bf16
                     const float*  __restrict__ wo,  // [H, H] row-major
                     float* __restrict__ out) {      // [M, H] fp32
  const int w    = threadIdx.x >> 5;
  const int lane = threadIdx.x & 31;
  const int m0   = blockIdx.x * 64  + (w >> 2) * 32;
  const int n0   = blockIdx.y * 128 + (w & 3) * 32;

  v8f c[2][2] = {};
  for (int k0 = 0; k0 < H_; k0 += 32) {
    v16bf a0 = load_a_bf16(a + (size_t)m0        * H_, H_, k0);
    v16bf a1 = load_a_bf16(a + (size_t)(m0 + 16) * H_, H_, k0);
    v16bf b0 = load_bT_f32(wo + (size_t)n0        * H_, H_, k0);
    v16bf b1 = load_bT_f32(wo + (size_t)(n0 + 16) * H_, H_, k0);
    c[0][0] = WMMA_BF16(a0, b0, c[0][0]);
    c[0][1] = WMMA_BF16(a0, b1, c[0][1]);
    c[1][0] = WMMA_BF16(a1, b0, c[1][0]);
    c[1][1] = WMMA_BF16(a1, b1, c[1][1]);
  }
#pragma unroll
  for (int mi = 0; mi < 2; ++mi)
#pragma unroll
    for (int t = 0; t < 2; ++t)
#pragma unroll
      for (int r = 0; r < 8; ++r) {
        const int m = m0 + mi * 16 + ((lane >> 4) << 3) + r;
        out[(size_t)m * H_ + n0 + t * 16 + (lane & 15)] = c[mi][t][r];
      }
}

extern "C" void kernel_launch(void* const* d_in, const int* in_sizes, int n_in,
                              void* d_out, int out_size, void* d_ws, size_t ws_size,
                              hipStream_t stream) {
  (void)in_sizes; (void)n_in; (void)out_size; (void)ws_size;
  const float* hs = (const float*)d_in[0];
  const int*   am = (const int*)  d_in[1];
  const float* wq = (const float*)d_in[2];
  const float* wk = (const float*)d_in[3];
  const float* wv = (const float*)d_in[4];
  const float* wo = (const float*)d_in[5];

  const size_t qkv_elems = (size_t)B_ * NH_ * S_ * HD_;   // 8,388,608
  __bf16* qb = (__bf16*)d_ws;          // [B,NH,S,HD]
  __bf16* kb = qb + qkv_elems;         // [B,NH,S,HD]
  __bf16* vb = kb + qkv_elems;         // [B,NH,HD,S]
  __bf16* ab = vb + qkv_elems;         // [B,S,H]  (attention output, bf16)

  dim3 blk(256);
  dim3 g1(M_ / 64, NH_);
  qkv_rope_kernel<<<g1, blk, 0, stream>>>(hs, wq, wk, wv, qb, kb, vb);

  dim3 g2(S_ / 128, NH_, B_);
  flash_attn_kernel<<<g2, blk, 0, stream>>>(qb, kb, vb, am, ab);

  dim3 g3(M_ / 64, H_ / 128);
  out_proj_kernel<<<g3, blk, 0, stream>>>(ab, wo, (float*)d_out);
}
